// SAGEConv1Layer_80547816669345
// MI455X (gfx1250) — matrix operand order repaired and checked by reading, not compiled, not run on hardware
//
#include <hip/hip_runtime.h>
#include <hip/hip_fp16.h>

typedef __attribute__((ext_vector_type(16))) _Float16 v16h;
typedef __attribute__((ext_vector_type(8)))  float    v8f;

#define F_DIM 64

// ------------------------------------------------------------------
// prep: pack per-node-type weight panels [64 x 16] in f16:
//   ind cols 0..4 = Wl[{0,3,6,7,10}], col5 = sum Wr over dst==ind rels
//   org cols 0..4 = Wl[{1,4,8,11,13}], col5 = sum Wr over dst==org rels
//   ext cols 0..3 = Wl[{2,5,9,12}]
// plus summed biases per dst type.
// ------------------------------------------------------------------
__global__ void prep_kernel(const float* __restrict__ Wl, const float* __restrict__ Wr,
                            const float* __restrict__ b, _Float16* __restrict__ Wp,
                            float* __restrict__ bsum) {
  const int ind_src[5] = {0,3,6,7,10};
  const int org_src[5] = {1,4,8,11,13};
  const int ext_src[4] = {2,5,9,12};
  const int ind_dst[7] = {0,1,2,7,8,9,13};
  const int org_dst[7] = {3,4,5,6,10,11,12};
  int k = threadIdx.x;
  if (k < F_DIM) {
    _Float16* wi = Wp + (0 * F_DIM + k) * 16;
    _Float16* wo = Wp + (1 * F_DIM + k) * 16;
    _Float16* we = Wp + (2 * F_DIM + k) * 16;
    for (int c = 0; c < 16; ++c) { wi[c] = (_Float16)0.f; wo[c] = (_Float16)0.f; we[c] = (_Float16)0.f; }
    for (int c = 0; c < 5; ++c) wi[c] = (_Float16)Wl[ind_src[c] * F_DIM + k];
    for (int c = 0; c < 5; ++c) wo[c] = (_Float16)Wl[org_src[c] * F_DIM + k];
    for (int c = 0; c < 4; ++c) we[c] = (_Float16)Wl[ext_src[c] * F_DIM + k];
    float si = 0.f, so = 0.f;
    for (int j = 0; j < 7; ++j) { si += Wr[ind_dst[j] * F_DIM + k]; so += Wr[org_dst[j] * F_DIM + k]; }
    wi[5] = (_Float16)si;
    wo[5] = (_Float16)so;
  }
  if (threadIdx.x == 0) {
    float s0 = 0.f, s1 = 0.f;
    for (int j = 0; j < 7; ++j) { s0 += b[ind_dst[j]]; s1 += b[org_dst[j]]; }
    bsum[0] = s0; bsum[1] = s1;
  }
}

// ------------------------------------------------------------------
// Projection GEMM via WMMA:  P[col][node] = sum_k x[node][k] * Wp[k][col]
// Each wave computes a 16(node) x 16(col) tile with two
// v_wmma_f32_16x16x32_f16 (K = 64).  8 waves / block -> 128 rows / block.
// P columns are padded to stride Np (multiple of 128), so C stores are
// ALWAYS two unconditional global_store_b128 per lane (col<NC mask only):
// C layout puts M = v + 8*(lane/16) in consecutive v -> 8 contiguous rows.
// ------------------------------------------------------------------
__global__ __launch_bounds__(256)
void proj_kernel(const float* __restrict__ x, const _Float16* __restrict__ Wp,
                 float* __restrict__ P, int N, int Np, int NC) {
  const int lane = threadIdx.x & 31;
  const int wave = threadIdx.x >> 5;
  const int tile = blockIdx.x * 128 + wave * 16;
  if (tile >= N) return;                // uniform per wave: EXEC stays all-ones
  const int m    = lane & 15;
  const int half = lane >> 4;
  int row = tile + m;
  if (row >= N) row = N - 1;            // clamp x-loads; pad rows never read back

  // A tile, 16-bit A 16x32 layout: lanes 0-15 hold K=0..15,
  // lanes 16-31 hold K=16..31 (contiguous, matching B).
  const float* xr = x + (size_t)row * F_DIM;
  v16h a0, a1;
#pragma unroll
  for (int j = 0; j < 8; ++j) {
    a0[j]     = (_Float16)xr[      16 * half + j];
    a0[j + 8] = (_Float16)xr[ 8 + 16 * half + j];
    a1[j]     = (_Float16)xr[32 + 16 * half + j];
    a1[j + 8] = (_Float16)xr[40 + 16 * half + j];
  }
  // B tile, 32x16: lane%16 = column, lanes 0-15 hold K=0..15, lanes 16-31 K=16..31.
  const int col = m;
  const int kb  = half * 16;
  v16h b0, b1;
#pragma unroll
  for (int j = 0; j < 16; ++j) {
    b0[j] = Wp[(kb + j) * 16 + col];
    b1[j] = Wp[(32 + kb + j) * 16 + col];
  }
  v8f c = {};
  c = __builtin_amdgcn_wmma_f32_16x16x32_f16(false, a0, false, b0, (short)0, c, false, false);
  c = __builtin_amdgcn_wmma_f32_16x16x32_f16(false, a1, false, b1, (short)0, c, false, false);

  if (col < NC) {
    float* pc = P + (size_t)col * Np + tile + 8 * half;   // 16B aligned: Np%128==0
    float4 lo = make_float4(c[0], c[1], c[2], c[3]);
    float4 hi = make_float4(c[4], c[5], c[6], c[7]);
    *(float4*)(pc)     = lo;          // global_store_b128
    *(float4*)(pc + 4) = hi;          // global_store_b128
  }
}

// ------------------------------------------------------------------
// Fused edge scatter for all 14 relations (blockIdx.y = relation):
// per edge, gather the 4B source projection and do two L2-resident
// hardware atomics (fp32 sum + u32 count).
// ------------------------------------------------------------------
struct EdgeParams {
  const int*    ei[14];
  const float*  p[14];
  float*        sum[14];
  unsigned int* cnt[14];
  int E;
};

__global__ __launch_bounds__(256)
void edge_all_kernel(EdgeParams ep) {
  const int r = blockIdx.y;
  const int e = blockIdx.x * blockDim.x + threadIdx.x;
  if (e >= ep.E) return;
  const int* __restrict__ ei = ep.ei[r];
  int src = ei[e];
  int dst = ei[ep.E + e];
  unsafeAtomicAdd(&ep.sum[r][dst], ep.p[r][src]);  // global_atomic_add_f32
  atomicAdd(&ep.cnt[r][dst], 1u);                  // global_atomic_add_u32
}

// ------------------------------------------------------------------
// Finalize: out = sigmoid( base_proj + sum_b + sum_r sum_r[n]/max(cnt_r[n],1) )
// sc layout per dst group: [rel][2][N] = {sum, cnt}
// ------------------------------------------------------------------
__global__ __launch_bounds__(256)
void finalize_kernel(const float* __restrict__ base_col, const float* __restrict__ bsum,
                     int bidx, const float* __restrict__ sc, int N, int nrels,
                     float* __restrict__ out) {
  int n = blockIdx.x * blockDim.x + threadIdx.x;
  if (n >= N) return;
  float acc = base_col[n] + bsum[bidx];
  const unsigned int* scu = (const unsigned int*)sc;
  for (int r = 0; r < nrels; ++r) {
    float s  = sc[(size_t)(2 * r) * N + n];
    float cf = (float)scu[(size_t)(2 * r + 1) * N + n];
    acc += s / fmaxf(cf, 1.0f);
  }
  out[n] = 1.0f / (1.0f + __expf(-acc));
}

// ------------------------------------------------------------------
extern "C" void kernel_launch(void* const* d_in, const int* in_sizes, int n_in,
                              void* d_out, int out_size, void* d_ws, size_t ws_size,
                              hipStream_t stream) {
  const float* x_ind = (const float*)d_in[0];
  const float* x_org = (const float*)d_in[1];
  const float* x_ext = (const float*)d_in[2];
  const float* Wl    = (const float*)d_in[3];
  const float* Wr    = (const float*)d_in[4];
  const float* b     = (const float*)d_in[5];
  const int*   ei    = (const int*)d_in[6];
  float* out = (float*)d_out;

  const int N_IND = in_sizes[0] / F_DIM;
  const int N_ORG = in_sizes[1] / F_DIM;
  const int N_EXT = in_sizes[2] / F_DIM;
  const int E     = in_sizes[6] / 28;       // [14, 2, E]

  // Padded column strides (multiple of the 128-row proj tile).
  const int NP_IND = (N_IND + 127) / 128 * 128;
  const int NP_ORG = (N_ORG + 127) / 128 * 128;
  const int NP_EXT = (N_EXT + 127) / 128 * 128;

  // ---- workspace layout (all regions 256B aligned) ----
  char* ws = (char*)d_ws;
  size_t off = 0;
  auto alignup = [](size_t v, size_t a) { return (v + a - 1) / a * a; };
  _Float16* Wp = (_Float16*)(ws + off); off = alignup(off + 3 * F_DIM * 16 * sizeof(_Float16), 256);
  float* bsum  = (float*)(ws + off);    off = alignup(off + 2 * sizeof(float), 256);
  float* P_ind = (float*)(ws + off);    off = alignup(off + (size_t)6 * NP_IND * 4, 256);
  float* P_org = (float*)(ws + off);    off = alignup(off + (size_t)6 * NP_ORG * 4, 256);
  float* P_ext = (float*)(ws + off);    off = alignup(off + (size_t)4 * NP_EXT * 4, 256);
  float* sc_ind = (float*)(ws + off);   off += (size_t)14 * N_IND * 4;   // 7 rels x {sum,cnt}
  float* sc_org = (float*)(ws + off);   off += (size_t)14 * N_ORG * 4;
  size_t sc_bytes = (size_t)14 * (N_IND + N_ORG) * 4;

  // ---- phase 0: pack weights, zero atomic tables ----
  prep_kernel<<<1, 64, 0, stream>>>(Wl, Wr, b, Wp, bsum);
  hipMemsetAsync(sc_ind, 0, sc_bytes, stream);

  // ---- phase 1: WMMA node-type projection GEMMs ----
  proj_kernel<<<(N_IND + 127) / 128, 256, 0, stream>>>(x_ind, Wp + 0 * F_DIM * 16, P_ind, N_IND, NP_IND, 6);
  proj_kernel<<<(N_ORG + 127) / 128, 256, 0, stream>>>(x_org, Wp + 1 * F_DIM * 16, P_org, N_ORG, NP_ORG, 6);
  proj_kernel<<<(N_EXT + 127) / 128, 256, 0, stream>>>(x_ext, Wp + 2 * F_DIM * 16, P_ext, N_EXT, NP_EXT, 4);

  // ---- phase 2: fused edge scatter over all 14 relations ----
  // {srctype, proj col, dsttype, rel-index-in-dst-group}
  struct RelInfo { int st, col, dt, ri; };
  const RelInfo rels[14] = {
    {0,0,0,0},{1,0,0,1},{2,0,0,2},{0,1,1,0},{1,1,1,1},{2,1,1,2},{0,2,1,3},
    {0,3,0,3},{1,2,0,4},{2,2,0,5},{0,4,1,4},{1,3,1,5},{2,3,1,6},{1,4,0,6}
  };
  const float* Psrc[3] = {P_ind, P_org, P_ext};
  const int    NPsrc[3] = {NP_IND, NP_ORG, NP_EXT};
  float*       scg[2]  = {sc_ind, sc_org};
  const int    Ndst[2] = {N_IND, N_ORG};

  EdgeParams ep;
  ep.E = E;
  for (int r = 0; r < 14; ++r) {
    const RelInfo& ri = rels[r];
    ep.ei[r]  = ei + (size_t)r * 2 * E;
    ep.p[r]   = Psrc[ri.st] + (size_t)ri.col * NPsrc[ri.st];
    ep.sum[r] = scg[ri.dt] + (size_t)(2 * ri.ri) * Ndst[ri.dt];
    ep.cnt[r] = (unsigned int*)(scg[ri.dt] + (size_t)(2 * ri.ri + 1) * Ndst[ri.dt]);
  }
  dim3 egrid((E + 255) / 256, 14);
  edge_all_kernel<<<egrid, 256, 0, stream>>>(ep);

  // ---- phase 3: combine + sigmoid ----
  finalize_kernel<<<(N_IND + 255) / 256, 256, 0, stream>>>(
      P_ind + (size_t)5 * NP_IND, bsum, 0, sc_ind, N_IND, 7, out);
  finalize_kernel<<<(N_ORG + 255) / 256, 256, 0, stream>>>(
      P_org + (size_t)5 * NP_ORG, bsum, 1, sc_org, N_ORG, 7, out + N_IND);
}
